// HippocampalComponent_73126113182061
// MI455X (gfx1250) — matrix operand order repaired and checked by reading, not compiled, not run on hardware
//
#include <hip/hip_runtime.h>

// ---------------------------------------------------------------------------
// CDNA5 (gfx1250) implementation of the hippocampal sparse-memory block.
//  - Dense GEMMs (x@W_down.T, x@W_gate.T) via v_wmma_f32_16x16x32_bf16 with
//    double-buffered LDS tiles filled by global_load_async_to_lds_b128
//    (ASYNCcnt) so tile copy overlaps the matrix pipe.
//  - Per K-step: all fragment ds_loads issued first (single dscnt wait),
//    then 4 back-to-back WMMAs.
//  - kWTA via exact bit-level binary search (relu => nonneg floats, uint order)
//  - Sparse stages (W_ca3, W_up) as gather-accumulate over pre-transposed
//    weights (coalesced row gathers, L2-resident)
// ---------------------------------------------------------------------------

typedef __attribute__((ext_vector_type(16))) __bf16 v16bf;
typedef __attribute__((ext_vector_type(8)))  float  v8f;

union Frag16 { v16bf v; uint4 q[2]; };

#define MAXNZ 128
#define NT 256
#define LDA 40   // padded LDS row stride in shorts (80B = 20 banks, conflict-free)

__device__ __forceinline__ unsigned short f32_to_bf16_rne(float f) {
  unsigned u = __float_as_uint(f);
  unsigned r = 0x7FFFu + ((u >> 16) & 1u);
  u += r;
  return (unsigned short)(u >> 16);
}

// CDNA5 async copy global -> LDS (GV mode, 16B), tracked by ASYNCcnt.
__device__ __forceinline__ void async_copy_b128(unsigned lds_byte, const void* gptr) {
  asm volatile("global_load_async_to_lds_b128 %0, %1, off"
               :: "v"(lds_byte), "v"((unsigned long long)(uintptr_t)gptr)
               : "memory");
}
__device__ __forceinline__ void wait_async0() {
  asm volatile("s_wait_asynccnt 0x0" ::: "memory");
}

// ---------------------------------------------------------------------------
// f32 -> bf16 conversion
// ---------------------------------------------------------------------------
__global__ __launch_bounds__(256) void cvt_f32_bf16_k(const float* __restrict__ s,
                                                      unsigned short* __restrict__ d, int n) {
  int i = blockIdx.x * 256 + threadIdx.x;
  if (i < n) d[i] = f32_to_bf16_rne(s[i]);
}

// ---------------------------------------------------------------------------
// 32x32 tiled transpose: D[c][r] = S[r][c], R,C multiples of 32
// ---------------------------------------------------------------------------
__global__ void transpose_f32_k(const float* __restrict__ S, float* __restrict__ D_,
                                int R, int C) {
  __shared__ float tile[32][33];
  int c0 = blockIdx.x * 32, r0 = blockIdx.y * 32;
  int tx = threadIdx.x, ty = threadIdx.y;
#pragma unroll
  for (int i = 0; i < 4; ++i) {
    int r = r0 + ty + i * 8;
    tile[ty + i * 8][tx] = S[(size_t)r * C + c0 + tx];
  }
  __syncthreads();
#pragma unroll
  for (int i = 0; i < 4; ++i) {
    int c = c0 + ty + i * 8;
    D_[(size_t)c * R + r0 + tx] = tile[tx][ty + i * 8];
  }
}

// ---------------------------------------------------------------------------
// NT GEMM: C[M,N] = op(A[M,K] @ B[N,K]^T), bf16 inputs, f32 accum via WMMA.
// mode 0: relu.  mode 1: sigmoid(v + bias[col]).
// Block: 256 thr = 8 waves; block tile 128(M) x 64(N), K-step 32.
// A/B tiles double-buffered in LDS, filled by async-to-LDS copies; each wave
// computes a 16x64 strip (A fragment reused across 4 WMMAs).
// Fragment layouts per CDNA5 ISA 7.12.2 (16-bit A 16x32 / B 32x16).
// ---------------------------------------------------------------------------
__global__ __launch_bounds__(256) void wmma_gemm_nt_k(
    const unsigned short* __restrict__ A, const unsigned short* __restrict__ B,
    float* __restrict__ C, const float* __restrict__ bias,
    int M, int N, int K, int mode) {
  __shared__ __attribute__((aligned(16))) unsigned short As[2][128 * LDA];
  __shared__ __attribute__((aligned(16))) unsigned short Bs[2][64 * LDA];

  const int tid  = threadIdx.x;
  const int wave = tid >> 5;
  const int lane = tid & 31;
  const int half = lane >> 4;   // hi/lo 16-lane group
  const int l16  = lane & 15;
  const int n0  = blockIdx.x * 64;
  const int m0b = blockIdx.y * 128;
  const int m0  = m0b + wave * 16;

  // --- async copy assignments: 16B chunks; A: 512 chunks (2/thread), B: 256 ---
  const int arow0 = tid >> 2;           // rows 0..63
  const int arow1 = arow0 + 64;         // rows 64..127
  const int kq    = tid & 3;            // which 8-short quarter of the 32-K tile
  const unsigned short* ag0 = A + (size_t)(m0b + arow0) * K + kq * 8;
  const unsigned short* ag1 = A + (size_t)(m0b + arow1) * K + kq * 8;
  const unsigned short* bg  = B + (size_t)(n0  + arow0) * K + kq * 8;
  const unsigned al0 = (unsigned)(uintptr_t)&As[0][arow0 * LDA + kq * 8];
  const unsigned al1 = (unsigned)(uintptr_t)&As[0][arow1 * LDA + kq * 8];
  const unsigned bl  = (unsigned)(uintptr_t)&Bs[0][arow0 * LDA + kq * 8];
  const unsigned aBufSz = (unsigned)sizeof(As[0]);
  const unsigned bBufSz = (unsigned)sizeof(Bs[0]);

  v8f acc[4] = {};
  const int nsteps = K >> 5;

  // prologue: fill buffer 0
  async_copy_b128(al0, ag0);
  async_copy_b128(al1, ag1);
  async_copy_b128(bl,  bg);
  wait_async0();
  __syncthreads();

  for (int s = 0; s < nsteps; ++s) {
    const int buf = s & 1;
    if (s + 1 < nsteps) {
      const int kb = (s + 1) << 5;
      const unsigned ao = (buf ^ 1) ? aBufSz : 0u;
      const unsigned bo = (buf ^ 1) ? bBufSz : 0u;
      async_copy_b128(al0 + ao, ag0 + kb);
      async_copy_b128(al1 + ao, ag1 + kb);
      async_copy_b128(bl  + bo, bg  + kb);
    }
    // --- compute on current buffer: issue ALL fragment ds_loads first ---
    Frag16 a;
    const int arow = (wave * 16 + l16) * LDA;
    a.q[0] = *(const uint4*)&As[buf][arow + half * 8];        // K = base+0..7
    a.q[1] = *(const uint4*)&As[buf][arow + half * 8 + 16];   // K = base+16..23
    Frag16 b[4];
#pragma unroll
    for (int t = 0; t < 4; ++t) {
      const int brow = (t * 16 + l16) * LDA;
      b[t].q[0] = *(const uint4*)&Bs[buf][brow + half * 8];      // K = half*8+0..7
      b[t].q[1] = *(const uint4*)&Bs[buf][brow + half * 8 + 16]; // K = half*8+16..23
    }
    // --- then 4 back-to-back WMMAs (independent accumulators, no hazards) ---
#pragma unroll
    for (int t = 0; t < 4; ++t) {
      acc[t] = __builtin_amdgcn_wmma_f32_16x16x32_bf16(
          false, a.v, false, b[t].v, (short)0, acc[t], false, false);
    }
    wait_async0();
    __syncthreads();
  }

  // D layout: VGPR r, lane L -> row = r + 8*(L>>4), col = L&15
#pragma unroll
  for (int t = 0; t < 4; ++t) {
    int col = n0 + t * 16 + l16;
    float bv = bias ? bias[col] : 0.0f;
#pragma unroll
    for (int r = 0; r < 8; ++r) {
      int row = m0 + half * 8 + r;
      float v = acc[t][r];
      if (mode == 0) v = fmaxf(v, 0.0f);
      else           v = 1.0f / (1.0f + __expf(-(v + bv)));
      C[(size_t)row * N + col] = v;
    }
  }
}

// ---------------------------------------------------------------------------
// Exact kWTA + L2-normalize + compaction on one LDS-resident row of nonneg f32.
// Threshold = k-th largest (binary search on float bit order); keeps
// v >= thresh (tie-inclusive, like the reference).
// ---------------------------------------------------------------------------
__device__ void kwta_compact(const float* srow, int N, int k,
                             int* redi, float* redf, int* cnt_sh,
                             int* out_idx, float* out_val, int* out_cnt) {
  const int tid = threadIdx.x;
  unsigned lo = 0u, hi = 0x7F800001u;
  while (hi - lo > 1u) {
    unsigned mid = lo + ((hi - lo) >> 1);
    int c = 0;
    for (int n = tid; n < N; n += NT)
      if (__float_as_uint(srow[n]) >= mid) c++;
    redi[tid] = c; __syncthreads();
    for (int s = 128; s > 0; s >>= 1) {
      if (tid < s) redi[tid] += redi[tid + s];
      __syncthreads();
    }
    int tot = redi[0]; __syncthreads();
    if (tot >= k) lo = mid; else hi = mid;
  }
  const unsigned tb = lo;

  float ss = 0.0f;
  for (int n = tid; n < N; n += NT) {
    float v = srow[n];
    if (__float_as_uint(v) >= tb) ss += v * v;
  }
  redf[tid] = ss; __syncthreads();
  for (int s = 128; s > 0; s >>= 1) {
    if (tid < s) redf[tid] += redf[tid + s];
    __syncthreads();
  }
  float inv = 1.0f / fmaxf(sqrtf(redf[0]), 1e-10f);
  if (tid == 0) *cnt_sh = 0;
  __syncthreads();
  for (int n = tid; n < N; n += NT) {
    float v = srow[n];
    if (__float_as_uint(v) >= tb) {
      int pos = atomicAdd(cnt_sh, 1);
      if (pos < MAXNZ) { out_idx[pos] = n; out_val[pos] = v * inv; }
    }
  }
  __syncthreads();
  if (tid == 0) *out_cnt = min(*cnt_sh, MAXNZ);
}

// ---------------------------------------------------------------------------
// Stage 1 kWTA over h1 rows -> compact (idx, val/norm) lists
// ---------------------------------------------------------------------------
__global__ __launch_bounds__(256) void kwta_rows_k(const float* __restrict__ H, int N,
                                                   const int* __restrict__ kp,
                                                   int* __restrict__ nz_idx,
                                                   float* __restrict__ nz_val,
                                                   int* __restrict__ nz_cnt) {
  __shared__ float srow[4096];
  __shared__ int   redi[NT];
  __shared__ float redf[NT];
  __shared__ int   cnt_sh;
  int m = blockIdx.x;
  const float* hr = H + (size_t)m * N;
  for (int n = threadIdx.x; n < N; n += NT) srow[n] = hr[n];
  __syncthreads();
  int k = max(1, min(kp[0], N));
  kwta_compact(srow, N, k, redi, redf, &cnt_sh,
               nz_idx + (size_t)m * MAXNZ, nz_val + (size_t)m * MAXNZ, nz_cnt + m);
}

// ---------------------------------------------------------------------------
// h2 = relu(sparse @ W_ca3.T) via gather over pre-transposed Wc3T, then
// fused kWTA+normalize; accumulates global sum for the validity scalar.
// ---------------------------------------------------------------------------
__global__ __launch_bounds__(256) void sparse_ca3_kwta_k(
    const int* __restrict__ nz_idx, const float* __restrict__ nz_val,
    const int* __restrict__ nz_cnt, const float* __restrict__ Wc3T,
    int N, const int* __restrict__ kp,
    int* __restrict__ s_idx, float* __restrict__ s_val, int* __restrict__ s_cnt,
    float* __restrict__ gsum) {
  __shared__ float srow[4096];
  __shared__ int   redi[NT];
  __shared__ float redf[NT];
  __shared__ int   cnt_sh;
  __shared__ int   lidx[MAXNZ];
  __shared__ float lval[MAXNZ];
  const int m = blockIdx.x, tid = threadIdx.x;
  const int cnt = nz_cnt[m];
  if (tid < cnt) {
    lidx[tid] = nz_idx[(size_t)m * MAXNZ + tid];
    lval[tid] = nz_val[(size_t)m * MAXNZ + tid];
  }
  __syncthreads();

  float acc[16];
#pragma unroll
  for (int i = 0; i < 16; ++i) acc[i] = 0.0f;
  for (int j = 0; j < cnt; ++j) {
    const float v = lval[j];
    const float* wr = Wc3T + (size_t)lidx[j] * N;
#pragma unroll
    for (int i = 0; i < 16; ++i) acc[i] += v * wr[tid + NT * i];
  }
  float lsum = 0.0f;
#pragma unroll
  for (int i = 0; i < 16; ++i) {
    float r = fmaxf(acc[i], 0.0f);
    srow[tid + NT * i] = r;
    lsum += r;
  }
  redf[tid] = lsum; __syncthreads();
  for (int s = 128; s > 0; s >>= 1) {
    if (tid < s) redf[tid] += redf[tid + s];
    __syncthreads();
  }
  if (tid == 0) atomicAdd(gsum, redf[0]);
  __syncthreads();
  int k = max(1, min(kp[0], N));
  kwta_compact(srow, N, k, redi, redf, &cnt_sh,
               s_idx + (size_t)m * MAXNZ, s_val + (size_t)m * MAXNZ, s_cnt + m);
}

// ---------------------------------------------------------------------------
// prediction = successor @ W_up.T (gather over WupT), out = x + g * pred.
// Validity fallback (successor := sparse) selected via global scalar.
// ---------------------------------------------------------------------------
__global__ __launch_bounds__(256) void pred_gate_out_k(
    const int* __restrict__ nz_idx, const float* __restrict__ nz_val,
    const int* __restrict__ nz_cnt,
    const int* __restrict__ s_idx, const float* __restrict__ s_val,
    const int* __restrict__ s_cnt,
    const float* __restrict__ WupT, const float* __restrict__ x,
    const float* __restrict__ gbuf, const float* __restrict__ gsum,
    float* __restrict__ out, int D) {
  __shared__ int   lidx[MAXNZ];
  __shared__ float lval[MAXNZ];
  const int m = blockIdx.x, tid = threadIdx.x;
  const bool valid = (*gsum >= 1e-10f);
  const int*   ip = valid ? s_idx : nz_idx;
  const float* vp = valid ? s_val : nz_val;
  const int   cnt = valid ? s_cnt[m] : nz_cnt[m];
  if (tid < cnt) {
    lidx[tid] = ip[(size_t)m * MAXNZ + tid];
    lval[tid] = vp[(size_t)m * MAXNZ + tid];
  }
  __syncthreads();
  float acc[4] = {0.f, 0.f, 0.f, 0.f};
  for (int j = 0; j < cnt; ++j) {
    const float v = lval[j];
    const float* wr = WupT + (size_t)lidx[j] * D;
#pragma unroll
    for (int i = 0; i < 4; ++i) acc[i] += v * wr[tid + NT * i];
  }
#pragma unroll
  for (int i = 0; i < 4; ++i) {
    size_t o = (size_t)m * D + tid + NT * i;
    out[o] = x[o] + gbuf[o] * acc[i];
  }
}

// ---------------------------------------------------------------------------
extern "C" void kernel_launch(void* const* d_in, const int* in_sizes, int n_in,
                              void* d_out, int out_size, void* d_ws, size_t ws_size,
                              hipStream_t stream) {
  const float* x   = (const float*)d_in[0];   // (8192, 1024)
  const float* Wd  = (const float*)d_in[1];   // (4096, 1024)
  const float* Wup = (const float*)d_in[2];   // (1024, 4096)
  const float* Wg  = (const float*)d_in[3];   // (1024, 1024)
  const float* bg  = (const float*)d_in[4];   // (1024,)
  const float* Wc3 = (const float*)d_in[5];   // (4096, 4096)
  const int*   kp  = (const int*)d_in[6];     // scalar k

  const int M = 4 * 2048, Dm = 1024, Nn = 4096;

  char* w = (char*)d_ws;
  size_t off = 0;
  auto alloc = [&](size_t bytes) -> void* {
    void* p = w + off;
    off = (off + bytes + 255) & ~(size_t)255;
    return p;
  };
  unsigned short* xb   = (unsigned short*)alloc((size_t)M * Dm * 2);
  unsigned short* Wdb  = (unsigned short*)alloc((size_t)Nn * Dm * 2);
  unsigned short* Wgb  = (unsigned short*)alloc((size_t)Dm * Dm * 2);
  float* h1    = (float*)alloc((size_t)M * Nn * 4);
  float* Wc3T  = (float*)alloc((size_t)Nn * Nn * 4);
  float* WupT  = (float*)alloc((size_t)Nn * Dm * 4);
  float* gbuf  = (float*)alloc((size_t)M * Dm * 4);
  int*   nzi   = (int*)alloc((size_t)M * MAXNZ * 4);
  float* nzv   = (float*)alloc((size_t)M * MAXNZ * 4);
  int*   nzc   = (int*)alloc((size_t)M * 4);
  int*   si    = (int*)alloc((size_t)M * MAXNZ * 4);
  float* sv    = (float*)alloc((size_t)M * MAXNZ * 4);
  int*   sc    = (int*)alloc((size_t)M * 4);
  float* gsum  = (float*)alloc(256);

  hipMemsetAsync(gsum, 0, sizeof(float), stream);

  // bf16 conversions for WMMA operands
  cvt_f32_bf16_k<<<(M * Dm + 255) / 256, 256, 0, stream>>>(x, xb, M * Dm);
  cvt_f32_bf16_k<<<(Nn * Dm + 255) / 256, 256, 0, stream>>>(Wd, Wdb, Nn * Dm);
  cvt_f32_bf16_k<<<(Dm * Dm + 255) / 256, 256, 0, stream>>>(Wg, Wgb, Dm * Dm);

  // pre-transpose sparse-gather weights for coalesced row gathers
  transpose_f32_k<<<dim3(Nn / 32, Nn / 32), dim3(32, 8), 0, stream>>>(Wc3, Wc3T, Nn, Nn);
  transpose_f32_k<<<dim3(Nn / 32, Dm / 32), dim3(32, 8), 0, stream>>>(Wup, WupT, Dm, Nn);

  // h1 = relu(x @ W_down.T)  [WMMA bf16, async-LDS double-buffered]
  wmma_gemm_nt_k<<<dim3(Nn / 64, M / 128), 256, 0, stream>>>(
      xb, Wdb, h1, nullptr, M, Nn, Dm, 0);
  // g = sigmoid(x @ W_gate.T + b_gate)  [WMMA bf16]
  wmma_gemm_nt_k<<<dim3(Dm / 64, M / 128), 256, 0, stream>>>(
      xb, Wgb, gbuf, bg, M, Dm, Dm, 1);

  // kWTA + normalize -> sparse lists
  kwta_rows_k<<<M, 256, 0, stream>>>(h1, Nn, kp, nzi, nzv, nzc);
  // successor = kwta(relu(sparse @ W_ca3.T)) normalized, + validity scalar
  sparse_ca3_kwta_k<<<M, 256, 0, stream>>>(nzi, nzv, nzc, Wc3T, Nn, kp, si, sv, sc, gsum);
  // out = x + g * (successor @ W_up.T)
  pred_gate_out_k<<<M, 256, 0, stream>>>(nzi, nzv, nzc, si, sv, sc, WupT, x, gbuf,
                                         gsum, (float*)d_out, Dm);
}
